// GaussianMixture_5385888989596
// MI455X (gfx1250) — compile-verified, matrix operand order checked by reading, not run on hardware
//
#include <hip/hip_runtime.h>

// ---------------------------------------------------------------------------
// GMM log-likelihood on gfx1250 (MI455X).
//   N=65536 points, K=32 components, F=128 features.
// Roofline: 2*N*K*F^2 = 68.7 GFLOP vs ~34 MB touched -> pure matrix-compute
// bound. GEMM via V_WMMA_F32_16X16X32_BF16 with hi/lo bf16 split (3 WMMAs per
// 16x16x32 chunk ~ f32 accuracy). B-operand (packed P, 2 MB) is staged into
// LDS by the Tensor Data Mover (double-buffered 2x32KB), cutting L2 read
// traffic 8x (all 8 waves of a block share the stream) and feeding the WMMAs
// from low-latency ds_load_b128 instead of latency-exposed global loads.
// ---------------------------------------------------------------------------

typedef __bf16 bf16_t;
typedef bf16_t v16bf __attribute__((ext_vector_type(16)));
typedef float  v8f   __attribute__((ext_vector_type(8)));
typedef unsigned int   u32x4v  __attribute__((ext_vector_type(4)));
typedef unsigned short u16x8v  __attribute__((ext_vector_type(8)));
typedef unsigned int   tdm_u32x4 __attribute__((ext_vector_type(4)));
typedef int            tdm_i32x8 __attribute__((ext_vector_type(8)));
typedef int            tdm_i32x4 __attribute__((ext_vector_type(4)));

#define NPTS  65536
#define KCOMP 32
#define FDIM  128
#define LOG2PI 1.8378770664093453f

// Packed B: 1024 fragments (k,gt,kc) x 2048 B. Within a fragment, four 512-B
// "planes": c=0 hi[0..7], c=1 hi[8..15], c=2 lo[0..7], c=3 lo[8..15]; each
// plane holds lane*16B chunks (8 bf16). LDS image is byte-identical.
#define FRAG_BYTES   2048
#define STAGE_BYTES  32768   // 16 fragments = half of one component
#define NSTAGES      64      // KCOMP * 2

union FragU { v16bf v; u32x4v q[2]; };

// Split f32 into bf16 hi (RNE) + bf16 lo (RNE of residual).
__device__ __forceinline__ void split_f32_bf16(float f, unsigned short &h,
                                               unsigned short &l) {
  unsigned u  = __float_as_uint(f);
  unsigned hb = (u + 0x7FFFu + ((u >> 16) & 1u)) >> 16;
  float    hf = __uint_as_float(hb << 16);
  float    lf = f - hf;
  unsigned ul = __float_as_uint(lf);
  unsigned lb = (ul + 0x7FFFu + ((ul >> 16) & 1u)) >> 16;
  h = (unsigned short)hb;
  l = (unsigned short)lb;
}

// ---------------------------------------------------------------------------
// TDM: 1-D contiguous copy of n8 x 8-byte elements, global -> LDS.
// D# per CDNA5 ISA 8.3/8.4: g0 = {count=1 | lds_addr | global_addr | type=2},
// g1 = {data_size=8B, tensor_dim0=n8, tensor_dim1=1, tile_dim0=n8,
//       tile_dim1/2=0 (unused), stride0=n8}. Groups 2/3 zero (<=2D tensor).
// ---------------------------------------------------------------------------
__device__ __forceinline__ void tdm_load_1d(unsigned int lds_off,
                                            unsigned long long gaddr,
                                            unsigned int n8) {
  tdm_u32x4 g0;
  g0[0] = 1u;                                             // count=1, user mode
  g0[1] = lds_off;                                        // lds_addr
  g0[2] = (unsigned int)(gaddr & 0xFFFFFFFFu);            // global_addr[31:0]
  g0[3] = (unsigned int)((gaddr >> 32) & 0x1FFFFFFu)      // global_addr[56:32]
          | (2u << 30);                                   // type=2 (image)
  tdm_i32x8 g1;
  g1[0] = (int)(3u << 16);                       // wg_mask=0, data_size=8B
  g1[1] = (int)((n8 & 0xFFFFu) << 16);           // tensor_dim0[15:0]
  g1[2] = (int)(((n8 >> 16) & 0xFFFFu)           // tensor_dim0[31:16]
                | (1u << 16));                   // tensor_dim1 = 1
  g1[3] = (int)((n8 & 0xFFFFu) << 16);           // tile_dim0 (16-bit)
  g1[4] = 0;                                     // tile_dim1=0, tile_dim2=0
  g1[5] = (int)n8;                               // tensor_dim0_stride[31:0]
  g1[6] = 0;
  g1[7] = 0;
  tdm_i32x4 z4 = {0, 0, 0, 0};
#if defined(__clang_major__) && (__clang_major__ >= 23)
  tdm_i32x8 z8 = {0, 0, 0, 0, 0, 0, 0, 0};
  __builtin_amdgcn_tensor_load_to_lds(g0, g1, z4, z4, z8, 0);
#else
  __builtin_amdgcn_tensor_load_to_lds(g0, g1, z4, z4, 0);
#endif
}

// ---------------------------------------------------------------------------
// Prep 1: pack P (K x F x F, col g contiguous) into fragment/plane layout,
// split into bf16 hi/lo. One thread per (fragment, lane) = 32768 threads.
// Fragment t = (k*8+gt)*4+kc. Lane: col N = lane&15; K set within chunk:
// lanes<16 -> {0..7,16..23}, lanes>=16 -> {8..15,24..31} (16-bit B layout).
// ---------------------------------------------------------------------------
__global__ __launch_bounds__(256) void gmm_pack_kernel(
    const float* __restrict__ P, bf16_t* __restrict__ Bp) {
  int t    = blockIdx.x * blockDim.x + threadIdx.x;  // 0..32767
  int lane = t & 31;
  int frag = t >> 5;
  int kc   = frag & 3;
  int gt   = (frag >> 2) & 7;
  int k    = frag >> 5;
  int g    = gt * 16 + (lane & 15);
  int fb   = kc * 32 + (lane >> 4) * 8;
  const float* Pk = P + (size_t)k * FDIM * FDIM;
  unsigned short hi[16], lo[16];
#pragma unroll
  for (int j = 0; j < 16; ++j) {
    int f = fb + ((j >> 3) * 16) + (j & 7);
    split_f32_bf16(Pk[(size_t)f * FDIM + g], hi[j], lo[j]);
  }
  unsigned short* base =
      (unsigned short*)Bp + (size_t)frag * 1024 + lane * 8;
  u16x8v c0, c1, c2, c3;
#pragma unroll
  for (int j = 0; j < 8; ++j) {
    c0[j] = hi[j];
    c1[j] = hi[j + 8];
    c2[j] = lo[j];
    c3[j] = lo[j + 8];
  }
  *(u16x8v*)(base + 0 * 256) = c0;
  *(u16x8v*)(base + 1 * 256) = c1;
  *(u16x8v*)(base + 2 * 256) = c2;
  *(u16x8v*)(base + 3 * 256) = c3;
}

// ---------------------------------------------------------------------------
// Prep 2: muP[k,g] = sum_f means[k,f]*P[k,f,g]; kconst[k] = logdet + log w
//         - 0.5*F*log(2pi).  K*F = 4096 threads.
// ---------------------------------------------------------------------------
__global__ __launch_bounds__(256) void gmm_muproj_kernel(
    const float* __restrict__ P, const float* __restrict__ means,
    const float* __restrict__ weights, float* __restrict__ muP,
    float* __restrict__ kconst) {
  int t = blockIdx.x * blockDim.x + threadIdx.x;
  int k = t >> 7;
  int g = t & 127;
  const float* Pk = P + (size_t)k * FDIM * FDIM;
  const float* mk = means + (size_t)k * FDIM;
  float acc = 0.0f;
  for (int f = 0; f < FDIM; ++f)
    acc = __builtin_fmaf(mk[f], Pk[(size_t)f * FDIM + g], acc);
  muP[t] = acc;
  if (g == 0) {
    float ld = 0.0f;
    for (int f = 0; f < FDIM; ++f) ld += __logf(Pk[(size_t)f * FDIM + f]);
    kconst[k] = ld + __logf(weights[k]) - 0.5f * (float)FDIM * LOG2PI;
  }
}

// ---------------------------------------------------------------------------
// Main kernel: 512 blocks x 8 waves; one wave per 16-row x-tile.
// TDM double-buffers 32-KB stages of packed P through LDS; all 8 waves
// consume each stage via ds_load_b128. Per stage: 48 WMMA + 64 ds loads.
// ---------------------------------------------------------------------------
__global__ __launch_bounds__(256) void gmm_main_kernel(
    const float* __restrict__ x, const bf16_t* __restrict__ Bp,
    const float* __restrict__ muP, const float* __restrict__ kconst,
    float* __restrict__ out) {
  __shared__ __align__(16) char smem[2 * STAGE_BYTES];

  const int lane  = threadIdx.x & 31;
  const int wave  = threadIdx.x >> 5;
  const int ntile = blockIdx.x * 8 + wave;
  const int row0  = ntile * 16;
  const int rl    = lane & 15;
  const int h16   = lane >> 4;

  const unsigned int lds_base =
      (unsigned int)(size_t)(void*)smem;  // LDS byte offset (low 32 bits)
  const unsigned long long gbase = (unsigned long long)(size_t)Bp;

  // ---- Build A fragments: 16x32 bf16 per K-chunk, hi + lo (reused 32K ways)
  v16bf ahi[4], alo[4];
  {
    const float* xrow = x + (size_t)(row0 + rl) * FDIM;
#pragma unroll
    for (int kc = 0; kc < 4; ++kc) {
      int fb = kc * 32 + h16 * 8;
      const float4 a0 = *(const float4*)(xrow + fb);
      const float4 a1 = *(const float4*)(xrow + fb + 4);
      const float4 a2 = *(const float4*)(xrow + fb + 16);
      const float4 a3 = *(const float4*)(xrow + fb + 20);
      float vals[16] = {a0.x, a0.y, a0.z, a0.w, a1.x, a1.y, a1.z, a1.w,
                        a2.x, a2.y, a2.z, a2.w, a3.x, a3.y, a3.z, a3.w};
      union { v16bf v; unsigned short u[16]; } uh, ul2;
#pragma unroll
      for (int j = 0; j < 16; ++j) split_f32_bf16(vals[j], uh.u[j], ul2.u[j]);
      ahi[kc] = uh.v;
      alo[kc] = ul2.v;
    }
  }

  // Prologue: wave 0 kicks off stage 0 into buffer 0.
  if (wave == 0) tdm_load_1d(lds_base, gbase, STAGE_BYTES / 8);

  float sq[8];
  float m[8], s[8];
#pragma unroll
  for (int v = 0; v < 8; ++v) {
    m[v] = -3.0e38f;
    s[v] = 0.0f;
  }

  for (int st = 0; st < NSTAGES; ++st) {
    const int k = st >> 1;
    const int h = st & 1;
    if (h == 0) {
#pragma unroll
      for (int v = 0; v < 8; ++v) sq[v] = 0.0f;
    }

    // Barrier A: every wave finished reading the buffer TDM will overwrite.
    __syncthreads();
    if (wave == 0) {
      if (st + 1 < NSTAGES) {
        tdm_load_1d(lds_base + (unsigned)((st + 1) & 1) * STAGE_BYTES,
                    gbase + (unsigned long long)(st + 1) * STAGE_BYTES,
                    STAGE_BYTES / 8);
        __builtin_amdgcn_s_wait_tensorcnt(1);  // stage st landed
      } else {
        __builtin_amdgcn_s_wait_tensorcnt(0);
      }
    }
    // Barrier B: stage st visible to all waves.
    __syncthreads();

    const char* buf = smem + (st & 1) * STAGE_BYTES;
#pragma unroll 2
    for (int gt2 = 0; gt2 < 4; ++gt2) {
      v8f acc = {};
#pragma unroll
      for (int kc = 0; kc < 4; ++kc) {
        const char* fr = buf + (gt2 * 4 + kc) * FRAG_BYTES + lane * 16;
        FragU fh, fl;
        fh.q[0] = *(const u32x4v*)(fr);
        fh.q[1] = *(const u32x4v*)(fr + 512);
        fl.q[0] = *(const u32x4v*)(fr + 1024);
        fl.q[1] = *(const u32x4v*)(fr + 1536);
        // x*P ~= xhi*Phi + xlo*Phi + xhi*Plo  (lo*lo dropped)
        acc = __builtin_amdgcn_wmma_f32_16x16x32_bf16(
            false, ahi[kc], false, fh.v, (short)0, acc, false, false);
        acc = __builtin_amdgcn_wmma_f32_16x16x32_bf16(
            false, alo[kc], false, fh.v, (short)0, acc, false, false);
        acc = __builtin_amdgcn_wmma_f32_16x16x32_bf16(
            false, ahi[kc], false, fl.v, (short)0, acc, false, false);
      }
      // D layout: lane = column (h*4+gt2)*16 + rl, VGPR v = row v + h16*8.
      float mu = muP[k * FDIM + (h * 4 + gt2) * 16 + rl];
#pragma unroll
      for (int v = 0; v < 8; ++v) {
        float d = acc[v] - mu;
        sq[v]   = __builtin_fmaf(d, d, sq[v]);
      }
    }

    if (h == 1) {
      // Reduce over the 16 columns within each lane-half.
#pragma unroll
      for (int v = 0; v < 8; ++v) {
        float t = sq[v];
        t += __shfl_xor(t, 1, 32);
        t += __shfl_xor(t, 2, 32);
        t += __shfl_xor(t, 4, 32);
        t += __shfl_xor(t, 8, 32);
        sq[v] = t;
      }
      float kc0 = kconst[k];
#pragma unroll
      for (int v = 0; v < 8; ++v) {
        float val = __builtin_fmaf(-0.5f, sq[v], kc0);
        float mo  = m[v];
        float mn  = fmaxf(mo, val);
        s[v]      = s[v] * __expf(mo - mn) + __expf(val - mn);
        m[v]      = mn;
      }
    }
  }

  if (rl == 0) {
#pragma unroll
    for (int v = 0; v < 8; ++v)
      out[row0 + h16 * 8 + v] = m[v] + __logf(s[v]);
  }
}

// ---------------------------------------------------------------------------
// Workspace layout (d_ws):
//   [0, 2 MiB)        Bp packed (1024 fragments x 2048 B)
//   [2 MiB, +16 KiB)  muP (K*F f32)
//   then              kconst (K f32)
// ---------------------------------------------------------------------------
extern "C" void kernel_launch(void* const* d_in, const int* in_sizes, int n_in,
                              void* d_out, int out_size, void* d_ws,
                              size_t ws_size, hipStream_t stream) {
  (void)in_sizes;
  (void)n_in;
  (void)out_size;
  (void)ws_size;
  const float* x     = (const float*)d_in[0];  // [N,F]
  const float* means = (const float*)d_in[1];  // [K,F]
  const float* P     = (const float*)d_in[2];  // [K,F,F]
  const float* w     = (const float*)d_in[3];  // [K]

  char*   ws     = (char*)d_ws;
  bf16_t* Bp     = (bf16_t*)ws;
  float*  muP    = (float*)(ws + (2u << 20));
  float*  kconst = (float*)(ws + (2u << 20) + KCOMP * FDIM * sizeof(float));

  gmm_pack_kernel<<<128, 256, 0, stream>>>(P, Bp);
  gmm_muproj_kernel<<<16, 256, 0, stream>>>(P, means, w, muP, kconst);
  gmm_main_kernel<<<512, 256, 0, stream>>>(x, Bp, muP, kconst,
                                           (float*)d_out);
}
